// AttributeGNN_18940805775405
// MI455X (gfx1250) — matrix operand order repaired and checked by reading, not compiled, not run on hardware
//
#include <hip/hip_runtime.h>

typedef __attribute__((ext_vector_type(16))) __bf16 v16bf;
typedef __attribute__((ext_vector_type(8)))  __bf16 v8bf;
typedef __attribute__((ext_vector_type(8)))  float  v8f;
typedef __attribute__((ext_vector_type(4)))  float  v4f;

#define DD 256
#define AA 16

// ---------------------------------------------------------------------------
// Prep: swizzle W_agg (f32 [256 x 512], weight[o,k]) into WMMA B-fragment
// layout, bf16.  GEMM1 computes agg = cat @ W^T, so B[k,n] = W[n,k]:
// fragment (kblk, nt), lane L holds column n = nt*16 + L%16, K values
// kblk*32 + 8*(L/16) + j + (j&8) (j=0..15)  ==  two 8-wide row chunks of W.
// ---------------------------------------------------------------------------
__global__ void prep_wagg_kernel(const float* __restrict__ W, __bf16* __restrict__ F) {
    int tid  = blockIdx.x * blockDim.x + threadIdx.x;   // 8192 threads
    int lane = tid & 31;
    int frag = tid >> 5;                                // kblk*16 + nt, 0..255
    int nt   = frag & 15;
    int kblk = frag >> 4;
    int o    = nt * 16 + (lane & 15);
    int hl   = (lane >> 4) & 1;
    int ks   = kblk * 32 + 8 * hl;
    __bf16* dst = F + (size_t)frag * 512 + (size_t)lane * 16;
    const float* src = W + (size_t)o * 512 + ks;
#pragma unroll
    for (int j = 0; j < 16; ++j) dst[j] = (__bf16)src[j + (j & 8)];
}

// ---------------------------------------------------------------------------
// Prep: swizzle P (f32 [A x 256 x 256], B[k,n] = P[a,k,n]) into B fragments.
// frag = a*128 + kblk*16 + nt ; lane L holds P[a, kblk*32+8*hl + j + (j&8), n].
// ---------------------------------------------------------------------------
__global__ void prep_p_kernel(const float* __restrict__ P, __bf16* __restrict__ F) {
    int tid  = blockIdx.x * blockDim.x + threadIdx.x;   // 65536 threads
    int lane = tid & 31;
    int frag = tid >> 5;                                // 0..2047
    int nt   = frag & 15;
    int kblk = (frag >> 4) & 7;
    int a    = frag >> 7;
    int n    = nt * 16 + (lane & 15);
    int hl   = (lane >> 4) & 1;
    int kb   = kblk * 32 + 8 * hl;
    __bf16* dst = F + (size_t)frag * 512 + (size_t)lane * 16;
    const float* src = P + ((size_t)a * DD + kb) * DD + n;
#pragma unroll
    for (int j = 0; j < 16; ++j) dst[j] = (__bf16)src[(size_t)(j + (j & 8)) * DD];
}

// Opaque pointer: hides loop-invariance from LICM so fragment loads are NOT
// hoisted out of the a-loop into scratch spills (round-3 pathology).
__device__ __forceinline__ const __bf16* opaque(const __bf16* p) {
    asm volatile("" : "+v"(p));
    return p;
}

// A-fragment from 16 f32 (two 8-wide chunks), streaming (non-temporal) loads.
__device__ __forceinline__ v16bf make_afrag_f32_nt(const float* c1, const float* c2) {
    const v4f* p1 = (const v4f*)c1;
    const v4f* p2 = (const v4f*)c2;
    v4f q0 = __builtin_nontemporal_load(p1);
    v4f q1 = __builtin_nontemporal_load(p1 + 1);
    v4f q2 = __builtin_nontemporal_load(p2);
    v4f q3 = __builtin_nontemporal_load(p2 + 1);
    float f[16] = {q0.x,q0.y,q0.z,q0.w, q1.x,q1.y,q1.z,q1.w,
                   q2.x,q2.y,q2.z,q2.w, q3.x,q3.y,q3.z,q3.w};
    v16bf af;
#pragma unroll
    for (int j = 0; j < 16; ++j) af[j] = (__bf16)f[j];
    return af;
}

__device__ __forceinline__ v16bf load_bfrag(const __bf16* p) {
    const v8bf* bp = (const v8bf*)p;
    v8bf b0 = bp[0], b1 = bp[1];
    return __builtin_shufflevector(b0, b1, 0,1,2,3,4,5,6,7,8,9,10,11,12,13,14,15);
}

// A-fragment from a row-major bf16 LDS tile (two ds_load_b128).
__device__ __forceinline__ v16bf lds_afrag(const __bf16* ls) {
    v8bf a0 = *(const v8bf*)ls;
    v8bf a1 = *(const v8bf*)(ls + 16);
    return __builtin_shufflevector(a0, a1, 0,1,2,3,4,5,6,7,8,9,10,11,12,13,14,15);
}

// ---------------------------------------------------------------------------
// Fused GNN kernel: one workgroup (512 thr / 16 waves) = 64 batch rows.
//   prolog : image tile [64x256] -> LDS (bf16), once
//   precomp: C_img = b_agg + x_tile @ W_src^T   (a-invariant!) -> registers
//   per a:
//     stage1: agg  = C_img + e_ba @ W_edge^T    (K=256)  -> LDS (bf16)
//     stage2: attr = agg @ P_fwd[a]             (K=256)  -> global + LDS
//     stage3: acc += relu(attr @ P_bwd[a])*sw[a](K=256)  -> registers
// wave w -> row-tile mt = w>>2, col tiles nt = (w&3)*4 .. +3.
// ---------------------------------------------------------------------------
__global__ __launch_bounds__(512) void gnn_fused_kernel(
    const float* __restrict__ img,  const float* __restrict__ edge,
    const float* __restrict__ bagg, const float* __restrict__ sw,
    const __bf16* __restrict__ waggF, const __bf16* __restrict__ pfwdF,
    const __bf16* __restrict__ pbwdF,
    float* __restrict__ attr_out, float* __restrict__ emb_out)
{
    __shared__ __align__(16) __bf16 s_img[64 * DD];  // 32 KB, read-only after fill
    __shared__ __align__(16) __bf16 sbuf[64 * DD];   // 32 KB, reused agg/attr

    const int tid  = threadIdx.x;
    const int lane = tid & 31;
    const int wave = tid >> 5;
    const int ln   = lane & 15;
    const int hl   = lane >> 4;
    const int mt   = wave >> 2;          // 0..3 (row tile)
    const int nt0  = (wave & 3) * 4;     // col tile base (4 tiles per wave)
    const int rowBase = blockIdx.x * 64 + mt * 16;
    const int rA   = rowBase + ln;       // global row this lane supplies to A
    const int lrow = mt * 16 + ln;       // LDS row for A fragments

    // ---- prolog: stage image tile to LDS as bf16 (once per block) ----
    {
        const size_t gbase = (size_t)blockIdx.x * 64 * DD;
#pragma unroll
        for (int it = 0; it < 4; ++it) {
            int off = (it * 512 + tid) * 8;      // 2048 chunks of 8 floats
            const v4f* p = (const v4f*)(img + gbase + off);
            v4f q0 = p[0], q1 = p[1];
            float f[8] = {q0.x,q0.y,q0.z,q0.w, q1.x,q1.y,q1.z,q1.w};
            v8bf v;
#pragma unroll
            for (int j = 0; j < 8; ++j) v[j] = (__bf16)f[j];
            *(v8bf*)(s_img + off) = v;
        }
    }
    __syncthreads();

    // ---- precompute a-invariant image-half of stage 1 (kept in registers) --
    v8f Cimg[4];
#pragma unroll
    for (int t = 0; t < 4; ++t) {
        float bv = bagg[(nt0 + t) * 16 + ln];
#pragma unroll
        for (int r = 0; r < 8; ++r) Cimg[t][r] = bv;
    }
    for (int kblk = 0; kblk < 8; ++kblk) {
        v16bf af = lds_afrag(s_img + (size_t)lrow * DD + kblk * 32 + 8 * hl);
        const __bf16* wb = waggF + ((size_t)(kblk * 16 + nt0) * 32 + lane) * 16;
#pragma unroll
        for (int t = 0; t < 4; ++t) {
            v16bf bf = load_bfrag(wb + (size_t)t * 512);
            Cimg[t] = __builtin_amdgcn_wmma_f32_16x16x32_bf16(
                false, af, false, bf, (short)0, Cimg[t], false, false);
        }
    }

    v8f acc[4];
#pragma unroll
    for (int t = 0; t < 4; ++t)
#pragma unroll
        for (int r = 0; r < 8; ++r) acc[t][r] = 0.0f;

    for (int a = 0; a < AA; ++a) {
        // ------- stage 1: agg = C_img + e_ba @ W_edge^T (edge half only) ----
        v8f C[4];
#pragma unroll
        for (int t = 0; t < 4; ++t) C[t] = Cimg[t];

        const float* erow = edge + ((size_t)rA * AA + a) * DD;
        if (a + 1 < AA)   // warm next edge row (uniform branch, EXEC stays full)
            __builtin_prefetch(edge + ((size_t)rA * AA + a + 1) * DD + hl * 128, 0, 1);
        // opaque base: stop LICM from hoisting these a-invariant loads
        const __bf16* waggE = opaque(waggF + (size_t)8 * 16 * 32 * 16);
        for (int kblk = 0; kblk < 8; ++kblk) {
            const float* src = erow + kblk * 32;
            v16bf af = make_afrag_f32_nt(src + 8 * hl, src + 16 + 8 * hl);
            const __bf16* wb = waggE + ((size_t)(kblk * 16 + nt0) * 32 + lane) * 16;
#pragma unroll
            for (int t = 0; t < 4; ++t) {
                v16bf bf = load_bfrag(wb + (size_t)t * 512);
                C[t] = __builtin_amdgcn_wmma_f32_16x16x32_bf16(
                    false, af, false, bf, (short)0, C[t], false, false);
            }
        }
#pragma unroll
        for (int t = 0; t < 4; ++t) {
            int col = (nt0 + t) * 16 + ln;
#pragma unroll
            for (int r = 0; r < 8; ++r)
                sbuf[(mt * 16 + r + 8 * hl) * DD + col] = (__bf16)C[t][r];
        }
        __syncthreads();

        // ---------------- stage 2: attr = agg @ P_fwd[a] ----------------
#pragma unroll
        for (int t = 0; t < 4; ++t)
#pragma unroll
            for (int r = 0; r < 8; ++r) C[t][r] = 0.0f;
        const __bf16* pfA = opaque(pfwdF + (size_t)a * 128 * 512);
        for (int kblk = 0; kblk < 8; ++kblk) {
            v16bf af = lds_afrag(sbuf + (size_t)lrow * DD + kblk * 32 + 8 * hl);
            const __bf16* wb = pfA + ((size_t)(kblk * 16 + nt0) * 32 + lane) * 16;
#pragma unroll
            for (int t = 0; t < 4; ++t) {
                v16bf bf = load_bfrag(wb + (size_t)t * 512);
                C[t] = __builtin_amdgcn_wmma_f32_16x16x32_bf16(
                    false, af, false, bf, (short)0, C[t], false, false);
            }
        }
        __syncthreads();   // partner waves finished reading agg before overwrite
#pragma unroll
        for (int t = 0; t < 4; ++t) {
            int col = (nt0 + t) * 16 + ln;
#pragma unroll
            for (int r = 0; r < 8; ++r) {
                int rg = rowBase + r + 8 * hl;
                __builtin_nontemporal_store(
                    C[t][r], attr_out + ((size_t)rg * AA + a) * DD + col);
                sbuf[(mt * 16 + r + 8 * hl) * DD + col] = (__bf16)C[t][r];
            }
        }
        __syncthreads();

        // ---------------- stage 3: acc += relu(attr @ P_bwd[a]) * sw[a] ----
#pragma unroll
        for (int t = 0; t < 4; ++t)
#pragma unroll
            for (int r = 0; r < 8; ++r) C[t][r] = 0.0f;
        const __bf16* pbA = opaque(pbwdF + (size_t)a * 128 * 512);
        for (int kblk = 0; kblk < 8; ++kblk) {
            v16bf af = lds_afrag(sbuf + (size_t)lrow * DD + kblk * 32 + 8 * hl);
            const __bf16* wb = pbA + ((size_t)(kblk * 16 + nt0) * 32 + lane) * 16;
#pragma unroll
            for (int t = 0; t < 4; ++t) {
                v16bf bf = load_bfrag(wb + (size_t)t * 512);
                C[t] = __builtin_amdgcn_wmma_f32_16x16x32_bf16(
                    false, af, false, bf, (short)0, C[t], false, false);
            }
        }
        float swa = sw[a];
#pragma unroll
        for (int t = 0; t < 4; ++t)
#pragma unroll
            for (int r = 0; r < 8; ++r)
                acc[t][r] += fmaxf(C[t][r], 0.0f) * swa;
        __syncthreads();   // before next iteration's stage-1 LDS writes
    }

#pragma unroll
    for (int t = 0; t < 4; ++t) {
        int col = (nt0 + t) * 16 + ln;
#pragma unroll
        for (int r = 0; r < 8; ++r)
            __builtin_nontemporal_store(
                acc[t][r], emb_out + (size_t)(rowBase + r + 8 * hl) * DD + col);
    }
}

extern "C" void kernel_launch(void* const* d_in, const int* in_sizes, int n_in,
                              void* d_out, int out_size, void* d_ws, size_t ws_size,
                              hipStream_t stream) {
    (void)n_in; (void)out_size; (void)ws_size;
    const float* img  = (const float*)d_in[0];   // [B, 256]
    const float* edge = (const float*)d_in[1];   // [B, 16, 256]
    const float* Wagg = (const float*)d_in[2];   // [256, 512]
    const float* bagg = (const float*)d_in[3];   // [256]
    const float* Pfwd = (const float*)d_in[4];   // [16, 256, 256]
    const float* Pbwd = (const float*)d_in[5];   // [16, 256, 256]
    const float* sw   = (const float*)d_in[6];   // [16]

    const int B = in_sizes[0] / DD;              // 16384

    float* attr_out = (float*)d_out;                         // [B, 16, 256]
    float* emb_out  = attr_out + (size_t)B * AA * DD;        // [B, 256]

    char* ws = (char*)d_ws;
    __bf16* waggF = (__bf16*)(ws);                           // 256 KB
    __bf16* pfwdF = (__bf16*)(ws + (256 << 10));             // 2 MB
    __bf16* pbwdF = (__bf16*)(ws + (256 << 10) + (2 << 20)); // 2 MB

    prep_wagg_kernel<<<32, 256, 0, stream>>>(Wagg, waggF);
    prep_p_kernel<<<256, 256, 0, stream>>>(Pfwd, pfwdF);
    prep_p_kernel<<<256, 256, 0, stream>>>(Pbwd, pbwdF);
    gnn_fused_kernel<<<B / 64, 512, 0, stream>>>(
        img, edge, bagg, sw, waggF, pfwdF, pbwdF, attr_out, emb_out);
}
